// CrossAttentionBlock_48842368090447
// MI455X (gfx1250) — matrix operand order repaired and checked by reading, not compiled, not run on hardware
//
#include <hip/hip_runtime.h>

// ---------------------------------------------------------------------------
// CDNA5 (gfx1250) cross-attention block.
// All GEMMs: V_WMMA_F32_16X16X32_F16 (f16 A/B, fp32 accumulate), wave32.
// Each wave computes a (16*TM)x(16*TN) register tile: TM A-fragments x
// TN B-fragments -> TM*TN WMMAs per K-slab (operand reuse, TM+TN fragment loads).
// MODE/TM/TN are compile-time so the K-loop is branch-free.
// ---------------------------------------------------------------------------

typedef __attribute__((ext_vector_type(16))) _Float16 v16h;
typedef __attribute__((ext_vector_type(8)))  float    v8f;

#define EPI_QHEAD   0   // scatter (row,col) -> qh[b,h,n,d] f16
#define EPI_KV      1   // scatter -> k[b,h,N,d] f16 and v_t[b,h,d,N] f16
#define EPI_SCORES  2   // fp32 * scale -> attn region of d_out
#define EPI_CTX     3   // A operand is fp32 (attn), out -> ctx[b,n,C] f16
#define EPI_PROJRES 4   // fp32: resid + acc + bias
#define EPI_GELU    5   // f16: exact gelu(acc + bias)
#define EPI_FFNRES  6   // fp32: resid + acc + bias -> d_out q region

struct GemmP {
  const _Float16* A16;   // f16 A, row-major [M x K], leading dim lda
  const float*    A32;   // fp32 A (EPI_CTX only)
  const _Float16* Bt;    // f16 B^T, row-major [Nc x K], leading dim ldb
  int  M, Nc, K, lda, ldb;
  long sA, sB;           // per-batch (blockIdx.z) element strides
  float scale;
  const float* bias;
  const float* resid;
  float*       outF32;
  _Float16*    outF16;
  _Float16*    outF16b;
  int n_seq, N_seq, dh, Cdim, heads;
};

union Frag { v16h v; unsigned u[8]; _Float16 h[16]; };

template <int MODE, int TM, int TN>
__global__ __launch_bounds__(128) void wmma_gemm_f16(GemmP p) {
  const int lane = threadIdx.x & 31;
  const int wv   = threadIdx.x >> 5;
  const int tileM0 = blockIdx.y * TM;              // first 16-row tile
  const int tileN0 = (blockIdx.x * 4 + wv) * TN;   // first 16-col tile of this wave
  const int z      = blockIdx.z;
  const int half   = lane >> 4;     // K-half selector (ISA 7.12.2 fragment layout)
  const int l15    = lane & 15;

  const _Float16* Arow16[TM];
  const float*    Arow32[TM];
#pragma unroll
  for (int t = 0; t < TM; ++t) {
    const int rowA = (tileM0 + t) * 16 + l15;
    if constexpr (MODE == EPI_CTX) {
      Arow32[t] = p.A32 + (size_t)z * p.sA + (size_t)rowA * p.lda;
      Arow16[t] = nullptr;
    } else {
      Arow16[t] = p.A16 + (size_t)z * p.sA + (size_t)rowA * p.lda;
      Arow32[t] = nullptr;
    }
  }
  const _Float16* Brow[TN];
#pragma unroll
  for (int t = 0; t < TN; ++t)
    Brow[t] = p.Bt + (size_t)z * p.sB + (size_t)((tileN0 + t) * 16 + l15) * p.ldb;

  v8f acc[TM][TN];
#pragma unroll
  for (int a = 0; a < TM; ++a)
#pragma unroll
    for (int b = 0; b < TN; ++b) acc[a][b] = v8f{};

  for (int k0 = 0; k0 < p.K; k0 += 32) {
    Frag af[TM];
    // A fragment 16x32 f16: lane row = M, VGPR i -> K = (i&4)*4 + half*8 + (i&3)*2
#pragma unroll
    for (int a = 0; a < TM; ++a) {
      if constexpr (MODE == EPI_CTX) {
#pragma unroll
        for (int i = 0; i < 8; ++i) {
          const int kk = k0 + ((i & 4) << 2) + half * 8 + (i & 3) * 2;
          const float2 t = *(const float2*)(Arow32[a] + kk);
          af[a].h[2 * i]     = (_Float16)t.x;
          af[a].h[2 * i + 1] = (_Float16)t.y;
        }
      } else {
#pragma unroll
        for (int i = 0; i < 8; ++i) {
          const int kk = k0 + ((i & 4) << 2) + half * 8 + (i & 3) * 2;
          af[a].u[i] = *(const unsigned*)(Arow16[a] + kk);
        }
      }
    }
    // B fragment 32x16 f16 (stored as B^T rows): VGPR j -> K = (j&4)*4 + half*8 + (j&3)*2
    Frag bf[TN];
#pragma unroll
    for (int t = 0; t < TN; ++t) {
#pragma unroll
      for (int j = 0; j < 8; ++j) {
        const int kk = k0 + ((j & 4) << 2) + half * 8 + (j & 3) * 2;
        bf[t].u[j] = *(const unsigned*)(Brow[t] + kk);
      }
    }
#pragma unroll
    for (int a = 0; a < TM; ++a)
#pragma unroll
      for (int b = 0; b < TN; ++b)
        acc[a][b] = __builtin_amdgcn_wmma_f32_16x16x32_f16(false, af[a].v, false, bf[b].v,
                                                           (short)0, acc[a][b], false, false);
  }

  // D layout: VGPR r holds (M = half*8 + r, N = lane&15)
#pragma unroll
  for (int a = 0; a < TM; ++a) {
#pragma unroll
    for (int b = 0; b < TN; ++b) {
      const int col = (tileN0 + b) * 16 + l15;
#pragma unroll
      for (int r = 0; r < 8; ++r) {
        const int   mrow = (tileM0 + a) * 16 + half * 8 + r;
        const float v    = acc[a][b][r];
        if constexpr (MODE == EPI_QHEAD) {
          const int bb = mrow / p.n_seq, i = mrow % p.n_seq;
          const int hd = col / p.dh,     j = col % p.dh;
          p.outF16[(((size_t)(bb * p.heads + hd)) * p.n_seq + i) * p.dh + j] = (_Float16)v;
        } else if constexpr (MODE == EPI_KV) {
          const int bb = mrow / p.N_seq, i = mrow % p.N_seq;
          const int c2 = (col < p.Cdim) ? col : col - p.Cdim;
          const int hd = c2 / p.dh, j = c2 % p.dh;
          if (col < p.Cdim)   // k: [b,h,N,d] row-major (natural B^T operand for QK^T)
            p.outF16 [(((size_t)(bb * p.heads + hd)) * p.N_seq + i) * p.dh + j] = (_Float16)v;
          else                // v stored transposed: [b,h,d,N] (B^T operand for attn@V)
            p.outF16b[(((size_t)(bb * p.heads + hd)) * p.dh + j) * p.N_seq + i] = (_Float16)v;
        } else if constexpr (MODE == EPI_SCORES) {
          p.outF32[(size_t)z * p.M * p.Nc + (size_t)mrow * p.Nc + col] = v * p.scale;
        } else if constexpr (MODE == EPI_CTX) {
          const int bb = z / p.heads, hd = z % p.heads;
          p.outF16[((size_t)(bb * p.n_seq + mrow)) * p.Cdim + hd * p.dh + col] = (_Float16)v;
        } else if constexpr (MODE == EPI_PROJRES) {
          const size_t o = (size_t)mrow * p.Nc + col;
          p.outF32[o] = p.resid[o] + v + p.bias[col];
        } else if constexpr (MODE == EPI_GELU) {
          const float tt = v + p.bias[col];
          p.outF16[(size_t)mrow * p.Nc + col] =
              (_Float16)(0.5f * tt * (1.0f + erff(tt * 0.70710678118654752f)));
        } else { // EPI_FFNRES
          const size_t o = (size_t)mrow * p.Nc + col;
          p.outF32[o] = p.resid[o] + v + p.bias[col];
        }
      }
    }
  }
}

template <int MODE, int TM, int TN>
static inline void launch_gemm(const GemmP& p, int batch, hipStream_t s) {
  // 4 waves per block, each wave a (16*TM)x(16*TN) tile.
  wmma_gemm_f16<MODE, TM, TN>
      <<<dim3(p.Nc / (64 * TN), p.M / (16 * TM), batch), dim3(128), 0, s>>>(p);
}

// ---------------- LayerNorm (fp32 in -> f16 out), one block per row ----------
__global__ void ln_to_f16(const float* __restrict__ x, const float* __restrict__ g,
                          const float* __restrict__ be, _Float16* __restrict__ out, int C) {
  const int row = blockIdx.x;
  const float* xr = x + (size_t)row * C;
  float s = 0.f, s2 = 0.f;
  for (int c = threadIdx.x; c < C; c += blockDim.x) {
    const float v = xr[c]; s += v; s2 += v * v;
  }
  __shared__ float shA[8], shB[8];
  const int lane = threadIdx.x & 31, wv = threadIdx.x >> 5;
  for (int o = 16; o > 0; o >>= 1) { s += __shfl_xor(s, o, 32); s2 += __shfl_xor(s2, o, 32); }
  if (lane == 0) { shA[wv] = s; shB[wv] = s2; }
  __syncthreads();
  if (wv == 0) {
    float a = (lane < 8) ? shA[lane] : 0.f;
    float b = (lane < 8) ? shB[lane] : 0.f;
    for (int o = 4; o > 0; o >>= 1) { a += __shfl_xor(a, o, 32); b += __shfl_xor(b, o, 32); }
    if (lane == 0) { shA[0] = a; shB[0] = b; }
  }
  __syncthreads();
  const float mean = shA[0] / C;
  const float var  = shB[0] / C - mean * mean;
  const float rstd = rsqrtf(var + 1e-5f);
  _Float16* orow = out + (size_t)row * C;
  for (int c = threadIdx.x; c < C; c += blockDim.x)
    orow[c] = (_Float16)((xr[c] - mean) * rstd * g[c] + be[c]);
}

// ---------------- in-place softmax over rows of length 2048, 256 threads -----
__global__ void softmax2048(float* __restrict__ a) {
  const int row = blockIdx.x;
  float* ar = a + (size_t)row * 2048;
  float vals[8];
  float mx = -3.402823466e38f;
#pragma unroll
  for (int i = 0; i < 8; ++i) { vals[i] = ar[threadIdx.x + i * 256]; mx = fmaxf(mx, vals[i]); }
  __shared__ float shA[8];
  const int lane = threadIdx.x & 31, wv = threadIdx.x >> 5;
  for (int o = 16; o > 0; o >>= 1) mx = fmaxf(mx, __shfl_xor(mx, o, 32));
  if (lane == 0) shA[wv] = mx;
  __syncthreads();
  if (wv == 0) {
    float m = (lane < 8) ? shA[lane] : -3.402823466e38f;
    for (int o = 4; o > 0; o >>= 1) m = fmaxf(m, __shfl_xor(m, o, 32));
    if (lane == 0) shA[0] = m;
  }
  __syncthreads();
  mx = shA[0];
  __syncthreads();
  float s = 0.f;
#pragma unroll
  for (int i = 0; i < 8; ++i) { vals[i] = expf(vals[i] - mx); s += vals[i]; }
  for (int o = 16; o > 0; o >>= 1) s += __shfl_xor(s, o, 32);
  if (lane == 0) shA[wv] = s;
  __syncthreads();
  if (wv == 0) {
    float t = (lane < 8) ? shA[lane] : 0.f;
    for (int o = 4; o > 0; o >>= 1) t += __shfl_xor(t, o, 32);
    if (lane == 0) shA[0] = t;
  }
  __syncthreads();
  const float inv = 1.0f / shA[0];
#pragma unroll
  for (int i = 0; i < 8; ++i) ar[threadIdx.x + i * 256] = vals[i] * inv;
}

// ---------------- elementwise fp32 -> f16 ------------------------------------
__global__ void f32_to_f16(const float* __restrict__ in, _Float16* __restrict__ out, long n) {
  const long i = (long)blockIdx.x * blockDim.x + threadIdx.x;
  if (i < n) out[i] = (_Float16)in[i];
}

// ---------------- weight transpose: fp32 [K x N] -> f16 [N x K] --------------
__global__ void transpose_w_f16(const float* __restrict__ in, _Float16* __restrict__ out,
                                int K, int N) {
  const long i = (long)blockIdx.x * blockDim.x + threadIdx.x;
  if (i >= (long)K * N) return;
  const int ncol = (int)(i / K);
  const int kk   = (int)(i % K);
  out[i] = (_Float16)in[(size_t)kk * N + ncol];
}

// ---------------------------------------------------------------------------
extern "C" void kernel_launch(void* const* d_in, const int* in_sizes, int n_in,
                              void* d_out, int out_size, void* d_ws, size_t ws_size,
                              hipStream_t stream) {
  const float* q    = (const float*)d_in[0];
  const float* x    = (const float*)d_in[1];
  const float* Wq   = (const float*)d_in[2];
  const float* Wkv  = (const float*)d_in[3];
  const float* Wprj = (const float*)d_in[4];
  const float* bprj = (const float*)d_in[5];
  const float* W1   = (const float*)d_in[6];
  const float* b1   = (const float*)d_in[7];
  const float* W2   = (const float*)d_in[8];
  const float* b2   = (const float*)d_in[9];
  const float* g1   = (const float*)d_in[10];
  const float* be1  = (const float*)d_in[11];
  const float* g2   = (const float*)d_in[12];
  const float* be2  = (const float*)d_in[13];

  constexpr int B = 4, n = 1024, N = 2048, C = 1024, H = 4096, HEADS = 16, D = 64;
  constexpr long BnC = (long)B * n * C;   // 4M
  constexpr long BNC = (long)B * N * C;   // 8M

  float* outQ = (float*)d_out;
  float* attn = outQ + BnC;               // [B,h,n,N] fp32, written in place

  // workspace carve-out (~152 MB total)
  char* w = (char*)d_ws;
  auto take = [&](size_t bytes) { char* p = w; w += (bytes + 255) & ~(size_t)255; return p; };
  _Float16* q16   = (_Float16*)take(BnC * 2);
  _Float16* xn16  = (_Float16*)take(BNC * 2);
  _Float16* Wqt   = (_Float16*)take((size_t)C * C * 2);
  _Float16* Wkvt  = (_Float16*)take((size_t)C * 2 * C * 2);
  _Float16* Wpt   = (_Float16*)take((size_t)C * C * 2);
  _Float16* W1t   = (_Float16*)take((size_t)C * H * 2);
  _Float16* W2t   = (_Float16*)take((size_t)H * C * 2);
  _Float16* qh16  = (_Float16*)take(BnC * 2);                       // [B,h,n,d]
  _Float16* k16   = (_Float16*)take((size_t)B * HEADS * N * D * 2); // [B,h,N,d]
  _Float16* vt16  = (_Float16*)take((size_t)B * HEADS * D * N * 2); // [B,h,d,N]
  _Float16* ctx16 = (_Float16*)take(BnC * 2);                       // [B,n,C]
  float*    q2    = (float*)   take(BnC * 4);                       // q + proj
  _Float16* ln216 = (_Float16*)take(BnC * 2);
  _Float16* hid16 = (_Float16*)take((size_t)B * n * H * 2);

  const dim3 b256(256);

  // 1) precision conversions / LN1 / weight transposes
  f32_to_f16<<<(unsigned)((BnC + 255) / 256), b256, 0, stream>>>(q, q16, BnC);
  ln_to_f16<<<B * N, b256, 0, stream>>>(x, g1, be1, xn16, C);
  transpose_w_f16<<<(C * C) / 256, b256, 0, stream>>>(Wq, Wqt, C, C);
  transpose_w_f16<<<(C * 2 * C) / 256, b256, 0, stream>>>(Wkv, Wkvt, C, 2 * C);
  transpose_w_f16<<<(C * C) / 256, b256, 0, stream>>>(Wprj, Wpt, C, C);
  transpose_w_f16<<<(C * H) / 256, b256, 0, stream>>>(W1, W1t, C, H);
  transpose_w_f16<<<(H * C) / 256, b256, 0, stream>>>(W2, W2t, H, C);

  // 2) qh = q @ Wq  (scatter per head)
  {
    GemmP p{};
    p.A16 = q16; p.Bt = Wqt; p.M = B * n; p.Nc = C; p.K = C; p.lda = C; p.ldb = C;
    p.outF16 = qh16; p.n_seq = n; p.dh = D; p.heads = HEADS;
    launch_gemm<EPI_QHEAD, 2, 2>(p, 1, stream);
  }
  // 3) kv = LN1(x) @ Wkv  (k row-major per head, v transposed per head)
  {
    GemmP p{};
    p.A16 = xn16; p.Bt = Wkvt; p.M = B * N; p.Nc = 2 * C; p.K = C; p.lda = C; p.ldb = C;
    p.outF16 = k16; p.outF16b = vt16;
    p.N_seq = N; p.dh = D; p.Cdim = C; p.heads = HEADS;
    launch_gemm<EPI_KV, 2, 2>(p, 1, stream);
  }
  // 4) scores = qh @ k^T * d^-0.5  -> fp32 straight into d_out attn region
  {
    GemmP p{};
    p.A16 = qh16; p.Bt = k16; p.M = n; p.Nc = N; p.K = D; p.lda = D; p.ldb = D;
    p.sA = (long)n * D; p.sB = (long)N * D;
    p.scale = 0.125f; p.outF32 = attn;
    launch_gemm<EPI_SCORES, 2, 2>(p, B * HEADS, stream);
  }
  // 5) softmax rows in place (this IS the attn output)
  softmax2048<<<B * HEADS * n, b256, 0, stream>>>(attn);
  // 6) ctx = attn @ v  (A = fp32 attn converted on load; B^T = v_t)
  {
    GemmP p{};
    p.A32 = attn; p.Bt = vt16; p.M = n; p.Nc = D; p.K = N; p.lda = N; p.ldb = N;
    p.sA = (long)n * N; p.sB = (long)D * N;
    p.outF16 = ctx16; p.n_seq = n; p.dh = D; p.Cdim = C; p.heads = HEADS;
    launch_gemm<EPI_CTX, 2, 1>(p, B * HEADS, stream);
  }
  // 7) q2 = q + ctx @ Wproj + bproj
  {
    GemmP p{};
    p.A16 = ctx16; p.Bt = Wpt; p.M = B * n; p.Nc = C; p.K = C; p.lda = C; p.ldb = C;
    p.bias = bprj; p.resid = q; p.outF32 = q2;
    launch_gemm<EPI_PROJRES, 2, 2>(p, 1, stream);
  }
  // 8) LN2(q2)
  ln_to_f16<<<B * n, b256, 0, stream>>>(q2, g2, be2, ln216, C);
  // 9) hid = gelu(LN2 @ W1 + b1)   (exact erf gelu)
  {
    GemmP p{};
    p.A16 = ln216; p.Bt = W1t; p.M = B * n; p.Nc = H; p.K = C; p.lda = C; p.ldb = C;
    p.bias = b1; p.outF16 = hid16;
    launch_gemm<EPI_GELU, 2, 2>(p, 1, stream);
  }
  // 10) out_q = q2 + hid @ W2 + b2
  {
    GemmP p{};
    p.A16 = hid16; p.Bt = W2t; p.M = B * n; p.Nc = C; p.K = H; p.lda = H; p.ldb = H;
    p.bias = b2; p.resid = q2; p.outF32 = outQ;
    launch_gemm<EPI_FFNRES, 2, 2>(p, 1, stream);
  }
}
